// Attention_71846212928150
// MI455X (gfx1250) — compile-verified, matrix operand order checked by reading, not run on hardware
//
#include <hip/hip_runtime.h>
#include <hip/hip_bf16.h>
#include <math.h>

typedef __attribute__((ext_vector_type(16))) _Float16 v16h;
typedef __attribute__((ext_vector_type(8)))  _Float16 v8h;
typedef __attribute__((ext_vector_type(4)))  _Float16 v4h;
typedef __attribute__((ext_vector_type(8)))  float    v8f;
typedef __attribute__((ext_vector_type(4)))  int      v4i;

#define BATCH 2
#define NTOK  2048
#define CH    1024
#define HEADS 16
#define DHEAD 64
#define ROWS  (BATCH*NTOK)   // 4096
#define INNER (HEADS*DHEAD)  // 1024
#define LN_EPS 1e-5f

// ---------------------------------------------------------------------------
// gfx1250 async global->LDS path (ASYNCcnt-tracked). Builtin signature
// (verified by compiler diagnostic): (int4 as1* src, int4 as3* dst, Ii, Ii).
// Guarded so a missing builtin degrades to a synchronous copy.
// ---------------------------------------------------------------------------
#if defined(__has_builtin)
#  if __has_builtin(__builtin_amdgcn_global_load_async_to_lds_b128)
#    define HAVE_ASYNC_LDS 1
#  endif
#endif
#ifndef HAVE_ASYNC_LDS
#  define HAVE_ASYNC_LDS 0
#endif

#if HAVE_ASYNC_LDS
#  if __has_builtin(__builtin_amdgcn_s_wait_asynccnt)
#    define WAIT_ASYNC(n) __builtin_amdgcn_s_wait_asynccnt(n)
#  else
#    define WAIT_ASYNC(n) asm volatile("s_wait_asynccnt %0" :: "i"(n) : "memory")
#  endif
#else
#  define WAIT_ASYNC(n) do {} while (0)
#endif

typedef __attribute__((address_space(1))) v4i as1_v4i;
typedef __attribute__((address_space(3))) v4i as3_v4i;

__device__ __forceinline__ void async_copy_b128(const _Float16* g, _Float16* l) {
#if HAVE_ASYNC_LDS
    __builtin_amdgcn_global_load_async_to_lds_b128((as1_v4i*)g, (as3_v4i*)l, 0, 0);
#else
    *(v8h*)l = *(const v8h*)g;
#endif
}

// ---------------------------------------------------------------------------
// WMMA fragment loader: 16x32 f16 A-fragment (also B-fragment when the source
// is stored transposed, i.e. [N,K] row-major). Per ISA 7.12.2: lane l holds
// row (l&15); lane-group g=l>>4 takes halves k0+8g..+7 (VGPR0-3) and
// k0+16+8g..+7 (VGPR4-7) -> two contiguous 16-byte loads per lane.
// ---------------------------------------------------------------------------
__device__ __forceinline__ v16h load_frag16x32(const _Float16* base, int stride,
                                               int k0, int lane) {
    int r = lane & 15;
    int g = lane >> 4;
    const _Float16* p = base + (size_t)r * stride + k0 + (g << 3);
    v8h lo = *(const v8h*)p;
    v8h hi = *(const v8h*)(p + 16);
    v16h a;
#pragma unroll
    for (int i = 0; i < 8; ++i) { a[i] = lo[i]; a[i + 8] = hi[i]; }
    return a;
}

__device__ __forceinline__ float silu_f(float y) {
    return y * (1.0f / (1.0f + __expf(-y)));
}

// ---------------------------------------------------------------------------
// Weight transpose + f16 cast:  w[K][N] f32 -> wT[N][K] f16  (LDS tiled)
// ---------------------------------------------------------------------------
__global__ void wtrans_f16_kernel(const float* __restrict__ w,
                                  _Float16* __restrict__ wT, int K, int N) {
    __shared__ float tile[32][33];
    int tx = threadIdx.x, ty = threadIdx.y;
    int n = blockIdx.x * 32 + tx;
    int kb = blockIdx.y * 32;
#pragma unroll
    for (int yy = 0; yy < 32; yy += 8)
        tile[ty + yy][tx] = w[(size_t)(kb + ty + yy) * N + n];
    __syncthreads();
    int k2 = kb + tx;
#pragma unroll
    for (int yy = 0; yy < 32; yy += 8) {
        int n2 = blockIdx.x * 32 + ty + yy;
        wT[(size_t)n2 * K + k2] = (_Float16)tile[tx][ty + yy];
    }
}

// ---------------------------------------------------------------------------
// s = silu(LayerNorm(x)) as f16.  One block (256 thr) per row of 1024.
// ---------------------------------------------------------------------------
__global__ void ln_silu_kernel(const float* __restrict__ x,
                               _Float16* __restrict__ s) {
    __shared__ float rs[8], rq[8];
    int row = blockIdx.x, t = threadIdx.x;
    const float4 v = ((const float4*)(x + (size_t)row * CH))[t];
    float sum = v.x + v.y + v.z + v.w;
    float ssq = v.x * v.x + v.y * v.y + v.z * v.z + v.w * v.w;
#pragma unroll
    for (int off = 16; off; off >>= 1) {
        sum += __shfl_xor(sum, off, 32);
        ssq += __shfl_xor(ssq, off, 32);
    }
    int wv = t >> 5, lane = t & 31;
    if (lane == 0) { rs[wv] = sum; rq[wv] = ssq; }
    __syncthreads();
    sum = 0.f; ssq = 0.f;
#pragma unroll
    for (int i = 0; i < 8; ++i) { sum += rs[i]; ssq += rq[i]; }
    float mean = sum * (1.0f / CH);
    float var  = ssq * (1.0f / CH) - mean * mean;
    float rstd = rsqrtf(var + LN_EPS);
    v4h o;
    o[0] = (_Float16)silu_f((v.x - mean) * rstd);
    o[1] = (_Float16)silu_f((v.y - mean) * rstd);
    o[2] = (_Float16)silu_f((v.z - mean) * rstd);
    o[3] = (_Float16)silu_f((v.w - mean) * rstd);
    ((v4h*)(s + (size_t)row * CH))[t] = o;
}

// ---------------------------------------------------------------------------
// WMMA GEMM: C[M,N] f32 = A[M,K] f16 * Bt[N,K]^T f16 + bias[N]
// Block = 256 thr (8 waves).  BM=128 (wave w -> rows w*16..), BN=64, BK=32.
// B tile (shared by all 8 waves) is double-buffered in LDS via async
// global->LDS copies; A fragments are register double-buffered.
// LDS row padded to 40 halves so ds_load_b128 fragment reads spread over
// 16 distinct 4-bank groups (no 4-way conflicts).
// ---------------------------------------------------------------------------
#define BK 32
#define BPAD 40

__global__ void gemm_f16_wmma_kernel(const _Float16* __restrict__ A,
                                     const _Float16* __restrict__ Bt,
                                     const float* __restrict__ bias,
                                     float* __restrict__ C,
                                     int M, int N, int K) {
    __shared__ _Float16 Blds[2][64 * BPAD];

    int tid = threadIdx.x;
    int wv = tid >> 5, lane = tid & 31;
    int m0 = blockIdx.x * 128 + wv * 16;
    int n0 = blockIdx.y * 64;

    // async staging assignment: thread t copies one 16B chunk:
    // B row n = t>>2 (64 rows), chunk = t&3 (4 x 8 halves = 32 halves)
    int bn = tid >> 2, chunk = tid & 3;
    const _Float16* gsrc = Bt + (size_t)(n0 + bn) * K + chunk * 8;
    _Float16* ldst[2] = { &Blds[0][bn * BPAD + chunk * 8],
                          &Blds[1][bn * BPAD + chunk * 8] };

    const _Float16* Abase = A + (size_t)m0 * K;

    // prologue: stage k0 = 0 into buffer 0, preload A fragment
    async_copy_b128(gsrc, ldst[0]);
    v16h a_cur = load_frag16x32(Abase, K, 0, lane);

    v8f acc0 = {}, acc1 = {}, acc2 = {}, acc3 = {};
    int cur = 0;

    for (int k0 = 0; k0 < K; k0 += BK) {
        bool havenext = (k0 + BK) < K;
        if (havenext) {
            async_copy_b128(gsrc + k0 + BK, ldst[cur ^ 1]);
            __builtin_prefetch(Abase + (size_t)(lane & 15) * K + k0 + BK, 0, 1);
        }
        v16h a_nxt = a_cur;
        if (havenext) a_nxt = load_frag16x32(Abase, K, k0 + BK, lane);

        if (havenext) { WAIT_ASYNC(1); } else { WAIT_ASYNC(0); }
        __syncthreads();                       // buf[cur] visible to all waves

        const _Float16* bb = &Blds[cur][0];
        v16h b0 = load_frag16x32(bb + 0 * 16 * BPAD, BPAD, 0, lane);
        v16h b1 = load_frag16x32(bb + 1 * 16 * BPAD, BPAD, 0, lane);
        v16h b2 = load_frag16x32(bb + 2 * 16 * BPAD, BPAD, 0, lane);
        v16h b3 = load_frag16x32(bb + 3 * 16 * BPAD, BPAD, 0, lane);
        acc0 = __builtin_amdgcn_wmma_f32_16x16x32_f16(false, a_cur, false, b0, (short)0, acc0, false, false);
        acc1 = __builtin_amdgcn_wmma_f32_16x16x32_f16(false, a_cur, false, b1, (short)0, acc1, false, false);
        acc2 = __builtin_amdgcn_wmma_f32_16x16x32_f16(false, a_cur, false, b2, (short)0, acc2, false, false);
        acc3 = __builtin_amdgcn_wmma_f32_16x16x32_f16(false, a_cur, false, b3, (short)0, acc3, false, false);

        __syncthreads();                       // reads done before overwrite
        a_cur = a_nxt;
        cur ^= 1;
    }

    int g = lane >> 4, cx = lane & 15;
#pragma unroll
    for (int r = 0; r < 8; ++r) {
        int row = m0 + r + 8 * g;
        float* crow = C + (size_t)row * N + n0 + cx;
        crow[ 0] = acc0[r] + bias[n0 + cx];
        crow[16] = acc1[r] + bias[n0 + 16 + cx];
        crow[32] = acc2[r] + bias[n0 + 32 + cx];
        crow[48] = acc3[r] + bias[n0 + 48 + cx];
    }
}

// ---------------------------------------------------------------------------
// Row-LayerNorm (affine) + scale, reshaped head-major f16 [B,H,N,D].
// ---------------------------------------------------------------------------
__global__ void ln_heads_kernel(const float* __restrict__ X,
                                const float* __restrict__ gam,
                                const float* __restrict__ bet,
                                _Float16* __restrict__ out, float scale) {
    __shared__ float rs[8], rq[8];
    int row = blockIdx.x, t = threadIdx.x;
    int b = row >> 11, n = row & 2047;
    const float4 v = ((const float4*)(X + (size_t)row * INNER))[t];
    float sum = v.x + v.y + v.z + v.w;
    float ssq = v.x * v.x + v.y * v.y + v.z * v.z + v.w * v.w;
#pragma unroll
    for (int off = 16; off; off >>= 1) {
        sum += __shfl_xor(sum, off, 32);
        ssq += __shfl_xor(ssq, off, 32);
    }
    int wv = t >> 5, lane = t & 31;
    if (lane == 0) { rs[wv] = sum; rq[wv] = ssq; }
    __syncthreads();
    sum = 0.f; ssq = 0.f;
#pragma unroll
    for (int i = 0; i < 8; ++i) { sum += rs[i]; ssq += rq[i]; }
    float mean = sum * (1.0f / INNER);
    float rstd = rsqrtf(ssq * (1.0f / INNER) - mean * mean + LN_EPS);
    int c = t * 4;
    int h = c >> 6, d = c & 63;
    float4 gv = ((const float4*)gam)[t];
    float4 bv = ((const float4*)bet)[t];
    v4h o;
    o[0] = (_Float16)(((v.x - mean) * rstd * gv.x + bv.x) * scale);
    o[1] = (_Float16)(((v.y - mean) * rstd * gv.y + bv.y) * scale);
    o[2] = (_Float16)(((v.z - mean) * rstd * gv.z + bv.z) * scale);
    o[3] = (_Float16)(((v.w - mean) * rstd * gv.w + bv.w) * scale);
    _Float16* dst = out + ((size_t)(b * HEADS + h) * NTOK + n) * DHEAD + d;
    *(v4h*)dst = o;
}

// ---------------------------------------------------------------------------
// v f32 [rows, inner] -> vt f16 [B,H,D,N]  (transposed per head for P*V)
// ---------------------------------------------------------------------------
__global__ void vcast_kernel(const float* __restrict__ V,
                             _Float16* __restrict__ vt) {
    int row = blockIdx.x, t = threadIdx.x;
    int b = row >> 11, n = row & 2047;
    const float4 v = ((const float4*)(V + (size_t)row * INNER))[t];
    float vv[4] = {v.x, v.y, v.z, v.w};
#pragma unroll
    for (int j = 0; j < 4; ++j) {
        int c = t * 4 + j;
        int h = c >> 6, d = c & 63;
        vt[((size_t)(b * HEADS + h) * DHEAD + d) * NTOK + n] = (_Float16)vv[j];
    }
}

// ---------------------------------------------------------------------------
// Attention: one block (8 waves) per (b,h, 16-query tile).
// LDS: S[16][2048] f32 | P[16][2048] f16 | rinv[16] | Oacc[2][4][16][16] f32
// ---------------------------------------------------------------------------
#define SMEM_S    0
#define SMEM_P    (16 * 2048 * 4)                       // 131072
#define SMEM_RINV (SMEM_P + 16 * 2048 * 2)              // 196608
#define SMEM_OACC (SMEM_RINV + 64)                      // 196672
#define SMEM_TOTAL (SMEM_OACC + 2 * 4 * 16 * 16 * 4)    // 204864

__global__ void attention_kernel(const _Float16* __restrict__ qh,
                                 const _Float16* __restrict__ kh,
                                 const _Float16* __restrict__ vt,
                                 _Float16* __restrict__ SO) {
    extern __shared__ char smem[];
    float*    S    = (float*)(smem + SMEM_S);
    _Float16* P    = (_Float16*)(smem + SMEM_P);
    float*    rinv = (float*)(smem + SMEM_RINV);
    float*    Oacc = (float*)(smem + SMEM_OACC);

    int tid = threadIdx.x;
    int lane = tid & 31, wv = tid >> 5;
    int bh = blockIdx.x >> 7;        // 128 query tiles per (b,h)
    int it = blockIdx.x & 127;
    int i0 = it * 16;

    const _Float16* qbase = qh + ((size_t)bh * NTOK + i0) * DHEAD;
    const _Float16* kbase = kh + (size_t)bh * NTOK * DHEAD;
    const _Float16* vbase = vt + (size_t)bh * DHEAD * NTOK;

    // ---- Phase 1: S = q_tile * k^T  (scale already folded into q) ----
    int g = lane >> 4, cx = lane & 15;
    v16h qf0 = load_frag16x32(qbase, DHEAD,  0, lane);   // loop-invariant
    v16h qf1 = load_frag16x32(qbase, DHEAD, 32, lane);
    for (int jt = wv; jt < 128; jt += 8) {
        const _Float16* kb = kbase + (size_t)jt * 16 * DHEAD;
        v16h b0 = load_frag16x32(kb, DHEAD,  0, lane);
        v16h b1 = load_frag16x32(kb, DHEAD, 32, lane);
        v8f acc = {};
        acc = __builtin_amdgcn_wmma_f32_16x16x32_f16(false, qf0, false, b0, (short)0, acc, false, false);
        acc = __builtin_amdgcn_wmma_f32_16x16x32_f16(false, qf1, false, b1, (short)0, acc, false, false);
        int col = jt * 16 + cx;
#pragma unroll
        for (int r = 0; r < 8; ++r)
            S[(size_t)(r + 8 * g) * NTOK + col] = acc[r];
    }
    __syncthreads();

    // ---- Phase 2: row softmax (exact, full row in LDS) ----
    {
        int row = tid >> 4, sub = tid & 15;
        const float* srow = S + (size_t)row * NTOK;
        float mx = -1e30f;
        for (int c = sub; c < NTOK; c += 16) mx = fmaxf(mx, srow[c]);
#pragma unroll
        for (int off = 8; off; off >>= 1) mx = fmaxf(mx, __shfl_xor(mx, off, 16));
        float sm = 0.f;
        _Float16* prow = P + (size_t)row * NTOK;
        for (int c = sub; c < NTOK; c += 16) {
            float e = __expf(srow[c] - mx);
            sm += e;
            prow[c] = (_Float16)e;
        }
#pragma unroll
        for (int off = 8; off; off >>= 1) sm += __shfl_xor(sm, off, 16);
        if (sub == 0) rinv[row] = 1.0f / sm;
    }
    __syncthreads();

    // ---- Phase 3: O = P * V, split-K across wave pairs ----
    {
        int dt = wv & 3;          // which 16-wide slice of D
        int ks = wv >> 2;         // K half: [0,1024) or [1024,2048)
        v8f acc = {};
        const _Float16* vb = vbase + (size_t)dt * 16 * NTOK;
        int kbeg = ks * 1024;
        for (int j = 0; j < 32; ++j) {
            int k0 = kbeg + j * 32;
            v16h a = load_frag16x32(P,  NTOK, k0, lane);
            v16h b = load_frag16x32(vb, NTOK, k0, lane);
            acc = __builtin_amdgcn_wmma_f32_16x16x32_f16(false, a, false, b, (short)0, acc, false, false);
        }
        float* op = Oacc + (size_t)((ks * 4 + dt) * 16) * 16;
#pragma unroll
        for (int r = 0; r < 8; ++r)
            op[(r + 8 * g) * 16 + cx] = acc[r];
    }
    __syncthreads();

    // ---- Phase 4: combine split-K, apply 1/sum, SiLU, store f16 ----
    {
        int b = bh >> 4, h = bh & 15;
        for (int e = tid; e < 16 * DHEAD; e += 256) {
            int rr = e >> 6, cc = e & 63;
            int dt = cc >> 4, c16 = cc & 15;
            float o = (Oacc[((0 * 4 + dt) * 16 + rr) * 16 + c16] +
                       Oacc[((1 * 4 + dt) * 16 + rr) * 16 + c16]) * rinv[rr];
            float so = silu_f(o);
            SO[(size_t)(b * NTOK + i0 + rr) * INNER + h * DHEAD + cc] = (_Float16)so;
        }
    }
}

// ---------------------------------------------------------------------------
extern "C" void kernel_launch(void* const* d_in, const int* in_sizes, int n_in,
                              void* d_out, int out_size, void* d_ws, size_t ws_size,
                              hipStream_t stream) {
    const float* x    = (const float*)d_in[0];
    const float* w_q  = (const float*)d_in[1];
    const float* b_q  = (const float*)d_in[2];
    const float* w_k  = (const float*)d_in[3];
    const float* b_k  = (const float*)d_in[4];
    const float* w_v  = (const float*)d_in[5];
    const float* b_v  = (const float*)d_in[6];
    const float* g_q  = (const float*)d_in[7];
    const float* be_q = (const float*)d_in[8];
    const float* g_k  = (const float*)d_in[9];
    const float* be_k = (const float*)d_in[10];
    const float* w_o  = (const float*)d_in[11];
    const float* b_o  = (const float*)d_in[12];
    float* out = (float*)d_out;

    char* ws = (char*)d_ws;
    const size_t SZ_ACT_H = (size_t)ROWS * CH * sizeof(_Float16);     // 8 MB
    const size_t SZ_W_H   = (size_t)CH * INNER * sizeof(_Float16);    // 2 MB
    const size_t SZ_ACT_F = (size_t)ROWS * INNER * sizeof(float);     // 16 MB

    size_t off = 0;
    _Float16* sX  = (_Float16*)(ws + off); off += SZ_ACT_H;
    _Float16* wqT = (_Float16*)(ws + off); off += SZ_W_H;
    _Float16* wkT = (_Float16*)(ws + off); off += SZ_W_H;
    _Float16* wvT = (_Float16*)(ws + off); off += SZ_W_H;
    _Float16* woT = (_Float16*)(ws + off); off += SZ_W_H;
    float*    Qr  = (float*)(ws + off);    size_t qr_off = off; off += SZ_ACT_F;
    float*    Kr  = (float*)(ws + off);    off += SZ_ACT_F;
    float*    Vr  = (float*)(ws + off);    off += SZ_ACT_F;
    _Float16* qh  = (_Float16*)(ws + off); off += SZ_ACT_H;
    _Float16* kh  = (_Float16*)(ws + off); off += SZ_ACT_H;
    _Float16* vt  = (_Float16*)(ws + off); off += SZ_ACT_H;
    _Float16* SO  = (_Float16*)(ws + qr_off);   // reuse Qr region (dead by then)

    dim3 tb(32, 8);
    dim3 tg(INNER / 32, CH / 32);
    wtrans_f16_kernel<<<tg, tb, 0, stream>>>(w_q, wqT, CH, INNER);
    wtrans_f16_kernel<<<tg, tb, 0, stream>>>(w_k, wkT, CH, INNER);
    wtrans_f16_kernel<<<tg, tb, 0, stream>>>(w_v, wvT, CH, INNER);
    wtrans_f16_kernel<<<tg, tb, 0, stream>>>(w_o, woT, INNER, CH);

    ln_silu_kernel<<<ROWS, 256, 0, stream>>>(x, sX);

    dim3 gg(ROWS / 128, INNER / 64);
    gemm_f16_wmma_kernel<<<gg, 256, 0, stream>>>(sX, wqT, b_q, Qr, ROWS, INNER, CH);
    gemm_f16_wmma_kernel<<<gg, 256, 0, stream>>>(sX, wkT, b_k, Kr, ROWS, INNER, CH);
    gemm_f16_wmma_kernel<<<gg, 256, 0, stream>>>(sX, wvT, b_v, Vr, ROWS, INNER, CH);

    const float qscale = 0.03125f;   // inner^-0.5 = 1/32
    ln_heads_kernel<<<ROWS, 256, 0, stream>>>(Qr, g_q, be_q, qh, qscale);
    ln_heads_kernel<<<ROWS, 256, 0, stream>>>(Kr, g_k, be_k, kh, 1.0f);
    vcast_kernel<<<ROWS, 256, 0, stream>>>(Vr, vt);

    attention_kernel<<<BATCH * HEADS * (NTOK / 16), 256, SMEM_TOTAL, stream>>>(qh, kh, vt, SO);

    dim3 go(ROWS / 128, CH / 64);
    gemm_f16_wmma_kernel<<<go, 256, 0, stream>>>(SO, woT, b_o, out, ROWS, CH, INNER);
}
